// Local_Feature_Extrection_35081292873869
// MI455X (gfx1250) — matrix operand order repaired and checked by reading, not compiled, not run on hardware
//
#include <hip/hip_runtime.h>
#include <hip/hip_bf16.h>
#include <math.h>

// ---------------------------------------------------------------------------
// Problem constants (from reference setup_inputs)
// ---------------------------------------------------------------------------
#define BB 8          // batch
#define NN 4096       // points per cloud
#define SS 1024       // NPOINT (fps samples)
#define NS 32         // NSAMPLE (knn)
#define DD 64         // input feature channels
#define P1 (BB*SS*NS) // 262144 grouped rows
#define P2 (BB*SS)    // 8192 centroid rows

typedef __attribute__((ext_vector_type(16))) _Float16 v16h;
typedef __attribute__((ext_vector_type(8)))  _Float16 v8h;
typedef __attribute__((ext_vector_type(8)))  float    v8f;

// ---------------------------------------------------------------------------
// 1. Farthest point sampling: one block per batch, 1024 threads (32 waves),
//    distances kept in registers, two-level argmax reduction.
// ---------------------------------------------------------------------------
__global__ __launch_bounds__(1024) void fps_kernel(const float* __restrict__ xyz,
                                                   int* __restrict__ fps_idx) {
  const int b = blockIdx.x;
  const int tid = threadIdx.x;
  const float* xb = xyz + (size_t)b * 3 * NN;

  float px[4], py[4], pz[4], dist[4];
#pragma unroll
  for (int i = 0; i < 4; ++i) {
    int j = tid + i * 1024;
    px[i] = xb[j];
    py[i] = xb[NN + j];
    pz[i] = xb[2 * NN + j];
    dist[i] = 1.0e10f;
  }

  __shared__ float rv[32];
  __shared__ int   ri[32];
  __shared__ int   sfar;
  if (tid == 0) sfar = 0;

  for (int it = 0; it < SS; ++it) {
    __syncthreads();
    int far = sfar;
    if (tid == 0) fps_idx[b * SS + it] = far;
    float cx = xb[far], cy = xb[NN + far], cz = xb[2 * NN + far];

    float bestv = -1.0f; int besti = 0;
#pragma unroll
    for (int i = 0; i < 4; ++i) {
      float dx = px[i] - cx, dy = py[i] - cy, dz = pz[i] - cz;
      float d = dx * dx + dy * dy + dz * dz;
      dist[i] = fminf(dist[i], d);
      if (dist[i] > bestv) { bestv = dist[i]; besti = tid + i * 1024; }
    }
    // wave32 argmax reduction
    for (int off = 16; off > 0; off >>= 1) {
      float ov = __shfl_down(bestv, off, 32);
      int   oi = __shfl_down(besti, off, 32);
      if (ov > bestv || (ov == bestv && oi < besti)) { bestv = ov; besti = oi; }
    }
    if ((tid & 31) == 0) { rv[tid >> 5] = bestv; ri[tid >> 5] = besti; }
    __syncthreads();
    if (tid < 32) {
      bestv = rv[tid]; besti = ri[tid];
      for (int off = 16; off > 0; off >>= 1) {
        float ov = __shfl_down(bestv, off, 32);
        int   oi = __shfl_down(besti, off, 32);
        if (ov > bestv || (ov == bestv && oi < besti)) { bestv = ov; besti = oi; }
      }
      if (tid == 0) sfar = besti;
    }
  }
}

// ---------------------------------------------------------------------------
// 2. Gather new_xyz (+ write transposed xyz output directly)
// ---------------------------------------------------------------------------
__global__ void gather_newxyz(const float* __restrict__ xyz, const int* __restrict__ fps_idx,
                              float* __restrict__ new_xyz, float* __restrict__ out0) {
  int i = blockIdx.x * blockDim.x + threadIdx.x;
  if (i >= P2) return;
  int b = i >> 10, s = i & 1023;
  int j = fps_idx[i];
#pragma unroll
  for (int c = 0; c < 3; ++c) {
    float v = xyz[((size_t)b * 3 + c) * NN + j];
    new_xyz[i * 3 + c] = v;
    out0[((size_t)b * 3 + c) * SS + s] = v;  // output tuple element 0: (B,3,S)
  }
}

// ---------------------------------------------------------------------------
// 3. KNN: one block (256 threads) per query; d2 in LDS, 33 iterative argmins
//    (first hit == the query point itself, dropped like reference idx[...,1:]).
// ---------------------------------------------------------------------------
__global__ __launch_bounds__(256) void knn_kernel(const float* __restrict__ xyz,
                                                  const float* __restrict__ new_xyz,
                                                  int* __restrict__ knn_idx) {
  const int bs = blockIdx.x;           // b*1024 + s
  const int b  = bs >> 10;
  const int tid = threadIdx.x;
  __shared__ float d2[NN];
  __shared__ float rv[8];
  __shared__ int   ri[8];
  const float* xb = xyz + (size_t)b * 3 * NN;
  float qx = new_xyz[bs * 3 + 0], qy = new_xyz[bs * 3 + 1], qz = new_xyz[bs * 3 + 2];

  for (int j = tid; j < NN; j += 256) {
    float dx = xb[j] - qx, dy = xb[NN + j] - qy, dz = xb[2 * NN + j] - qz;
    d2[j] = dx * dx + dy * dy + dz * dz;
  }
  __syncthreads();

  for (int t = 0; t < NS + 1; ++t) {
    float bv = 3.4e38f; int bi = NN;
    for (int j = tid; j < NN; j += 256) {
      float v = d2[j];
      if (v < bv || (v == bv && j < bi)) { bv = v; bi = j; }
    }
    for (int off = 16; off > 0; off >>= 1) {
      float ov = __shfl_down(bv, off, 32);
      int   oi = __shfl_down(bi, off, 32);
      if (ov < bv || (ov == bv && oi < bi)) { bv = ov; bi = oi; }
    }
    if ((tid & 31) == 0) { rv[tid >> 5] = bv; ri[tid >> 5] = bi; }
    __syncthreads();
    if (tid == 0) {
      for (int w = 1; w < 8; ++w)
        if (rv[w] < bv || (rv[w] == bv && ri[w] < bi)) { bv = rv[w]; bi = ri[w]; }
      d2[bi] = 3.0e38f;
      if (t > 0) knn_idx[(size_t)bs * NS + (t - 1)] = bi;
    }
    __syncthreads();
  }
}

// ---------------------------------------------------------------------------
// 4. Transpose points (B,64,N) f32 -> (B,N,64) f16 for contiguous gathers
// ---------------------------------------------------------------------------
__global__ void transpose_points(const float* __restrict__ points, _Float16* __restrict__ pts_t) {
  int i = blockIdx.x * blockDim.x + threadIdx.x;
  if (i >= BB * DD * NN) return;
  int n = i % NN; int r = i / NN; int c = r % DD; int b = r / DD;
  pts_t[((size_t)b * NN + n) * DD + c] = (_Float16)points[i];
}

// ---------------------------------------------------------------------------
// 5. Build grouped geometry, aug GEMM input (10 ch padded->32), and the
//    points part of the new_points GEMM input (96-ch rows, channels 0..63).
// ---------------------------------------------------------------------------
__global__ void build_group(const float* __restrict__ xyz, const _Float16* __restrict__ pts_t,
                            const float* __restrict__ new_xyz, const int* __restrict__ knn_idx,
                            float* __restrict__ grouped, float* __restrict__ gdist,
                            _Float16* __restrict__ aug_in, _Float16* __restrict__ np_in) {
  int p = blockIdx.x * blockDim.x + threadIdx.x;
  if (p >= P1) return;
  int bs = p >> 5; int b = bs >> 10;
  int j = knn_idx[p];
  const float* xb = xyz + (size_t)b * 3 * NN;
  float gx = xb[j], gy = xb[NN + j], gz = xb[2 * NN + j];
  float cx = new_xyz[bs * 3], cy = new_xyz[bs * 3 + 1], cz = new_xyz[bs * 3 + 2];
  float dx = gx - cx, dy = gy - cy, dz = gz - cz;
  float nx = dx * dx, ny = dy * dy, nz = dz * dz;
  float dd = sqrtf(nx + ny + nz);
  grouped[p * 3 + 0] = gx; grouped[p * 3 + 1] = gy; grouped[p * 3 + 2] = gz;
  gdist[p] = dd;
  _Float16* ar = aug_in + (size_t)p * 32;
  ar[0] = (_Float16)cx; ar[1] = (_Float16)cy; ar[2] = (_Float16)cz;
  ar[3] = (_Float16)gx; ar[4] = (_Float16)gy; ar[5] = (_Float16)gz;
  ar[6] = (_Float16)nx; ar[7] = (_Float16)ny; ar[8] = (_Float16)nz;
  ar[9] = (_Float16)dd;
#pragma unroll
  for (int k = 10; k < 32; ++k) ar[k] = (_Float16)0.0f;
  const _Float16* src = pts_t + ((size_t)b * NN + j) * DD;
  _Float16* nr = np_in + (size_t)p * 96;
#pragma unroll 8
  for (int c = 0; c < DD; ++c) nr[c] = src[c];
}

__global__ void fill_fp_points(const _Float16* __restrict__ pts_t, const int* __restrict__ fps_idx,
                               _Float16* __restrict__ fp_in) {
  int i = blockIdx.x * blockDim.x + threadIdx.x;
  if (i >= P2 * DD) return;
  int c = i % DD; int bs = i / DD; int b = bs >> 10;
  int j = fps_idx[bs];
  fp_in[(size_t)bs * 96 + c] = pts_t[((size_t)b * NN + j) * DD + c];
}

// max over neighbors of normalized aug (already stored at np_in[...,64..95])
__global__ void maxout_fp(const _Float16* __restrict__ np_in, _Float16* __restrict__ fp_in) {
  int i = blockIdx.x * blockDim.x + threadIdx.x;
  if (i >= P2 * 32) return;
  int k = i & 31; int bs = i >> 5;
  float m = -3.4e38f;
  for (int n = 0; n < NS; ++n)
    m = fmaxf(m, (float)np_in[((size_t)(bs * NS + n)) * 96 + 64 + k]);
  fp_in[(size_t)bs * 96 + 64 + k] = (_Float16)m;
}

// ---------------------------------------------------------------------------
// 6. WMMA GEMM with fused BN statistics:
//      Y16[P x Cout] = A[P x K] * W[Cout x K]^T + bias      (f16 out, pre-BN)
//      gsum/gsumsq[c] += per-column sums (exact, from f32 accumulators)
//    One wave computes 16 rows x all Cout columns; 8 waves per block.
//    Grids are exact (P % 128 == 0) so block-wide barriers are safe.
// ---------------------------------------------------------------------------
template<int KDIM, int NOB>
__global__ __launch_bounds__(256) void gemm_wmma(const _Float16* __restrict__ A,
                                                 const _Float16* __restrict__ W,
                                                 const float* __restrict__ bias,
                                                 _Float16* __restrict__ Y,
                                                 float* __restrict__ gsum,
                                                 float* __restrict__ gsumsq) {
  const int Cout = NOB * 16;
  __shared__ float lsum[128];
  __shared__ float lsq[128];
  if (threadIdx.x < Cout) { lsum[threadIdx.x] = 0.0f; lsq[threadIdx.x] = 0.0f; }
  __syncthreads();

  int wave = (int)((blockIdx.x * blockDim.x + threadIdx.x) >> 5);
  int lane = threadIdx.x & 31;
  int row0 = wave << 4;
  int m = lane & 15;
  int half = lane >> 4;

  const _Float16* arow = A + (size_t)(row0 + m) * KDIM;
  v8f acc[NOB];
#pragma unroll
  for (int i = 0; i < NOB; ++i) acc[i] = (v8f)0.0f;

#pragma unroll
  for (int kb = 0; kb < KDIM; kb += 32) {
    // 16-bit A 16x32 fragment: two contiguous 16B loads per lane
    v8h alo = *(const v8h*)(arow + kb + half * 16);
    v8h ahi = *(const v8h*)(arow + kb + half * 16 + 8);
    if (kb + 32 < KDIM) __builtin_prefetch(arow + kb + 32, 0, 0);  // global_prefetch_b8
    v16h a;
#pragma unroll
    for (int e = 0; e < 8; ++e) { a[e] = alo[e]; a[e + 8] = ahi[e]; }
#pragma unroll
    for (int ob = 0; ob < NOB; ++ob) {
      // 16-bit B 32x16 fragment: one contiguous 32B load per lane
      v16h bf = *(const v16h*)(W + (size_t)(ob * 16 + m) * KDIM + kb + half * 16);
      acc[ob] = __builtin_amdgcn_wmma_f32_16x16x32_f16(
          false, a, false, bf, (short)0, acc[ob], false, false);
    }
  }

  // epilogue: bias add, f16 store, exact per-column stats
#pragma unroll
  for (int ob = 0; ob < NOB; ++ob) {
    int col = ob * 16 + m;
    float bv = bias[col];
    _Float16* yr = Y + (size_t)(row0 + half * 8) * Cout + col;
    float s = 0.0f, s2 = 0.0f;
#pragma unroll
    for (int r = 0; r < 8; ++r) {
      float v = acc[ob][r] + bv;
      s += v; s2 += v * v;
      yr[(size_t)r * Cout] = (_Float16)v;
    }
    // combine the two half-tiles (lane ^ 16) so lanes 0..15 own full columns
    s  += __shfl_xor(s, 16, 32);
    s2 += __shfl_xor(s2, 16, 32);
    if (half == 0) {
      atomicAdd(&lsum[col], s);     // ds_add_f32
      atomicAdd(&lsq[col], s2);
    }
  }
  __syncthreads();
  if (threadIdx.x < Cout) {
    atomicAdd(&gsum[threadIdx.x], lsum[threadIdx.x]);
    atomicAdd(&gsumsq[threadIdx.x], lsq[threadIdx.x]);
  }
}

// ---------------------------------------------------------------------------
// 7. BatchNorm helpers (batch stats over all rows, per channel)
// ---------------------------------------------------------------------------
__global__ void zero_kernel(float* __restrict__ p, int n) {
  int i = blockIdx.x * blockDim.x + threadIdx.x;
  if (i < n) p[i] = 0.0f;
}

__global__ void bn_finalize(const float* __restrict__ sum, const float* __restrict__ sumsq,
                            const float* __restrict__ gamma, const float* __restrict__ beta,
                            float* __restrict__ scale, float* __restrict__ shift,
                            int C, float invPn) {
  int c = blockIdx.x * blockDim.x + threadIdx.x;
  if (c >= C) return;
  float mean = sum[c] * invPn;
  float var  = sumsq[c] * invPn - mean * mean;
  float sc = gamma[c] * rsqrtf(var + 1e-5f);
  scale[c] = sc;
  shift[c] = beta[c] - mean * sc;
}

// normalize + ReLU, f16 in -> f16 out into (possibly strided) destination rows
__global__ void bn_apply_f16(const _Float16* __restrict__ Y, const float* __restrict__ scale,
                             const float* __restrict__ shift, _Float16* __restrict__ out,
                             int P, int C, int ostride, int ooff) {
  size_t i = (size_t)blockIdx.x * blockDim.x + threadIdx.x;
  if (i >= (size_t)P * C) return;
  int c = (int)(i % C); size_t p = i / C;
  float v = (float)Y[i] * scale[c] + shift[c];
  v = fmaxf(v, 0.0f);
  out[p * ostride + ooff + c] = (_Float16)v;
}

// ---------------------------------------------------------------------------
// 8. Build LAA GEMM input rows: [centre(3) | grouped(3) | gdist(1) | delta_f(128) | pad->160]
// ---------------------------------------------------------------------------
__global__ __launch_bounds__(160) void build_laa(const float* __restrict__ new_xyz,
                                                 const float* __restrict__ grouped,
                                                 const float* __restrict__ gdist,
                                                 const _Float16* __restrict__ np_final,
                                                 const _Float16* __restrict__ centre_feat,
                                                 _Float16* __restrict__ laa_in) {
  int p = blockIdx.x;
  int t = threadIdx.x;
  int bs = p >> 5;
  _Float16 v;
  if (t < 3)        v = (_Float16)new_xyz[bs * 3 + t];
  else if (t < 6)   v = (_Float16)grouped[p * 3 + (t - 3)];
  else if (t == 6)  v = (_Float16)gdist[p];
  else if (t < 135) {
    int c = t - 7;
    v = (_Float16)((float)np_final[(size_t)p * 128 + c] -
                   (float)centre_feat[(size_t)bs * 128 + c]);
  } else v = (_Float16)0.0f;
  laa_in[(size_t)p * 160 + t] = v;
}

// ---------------------------------------------------------------------------
// 9. Fused BN+ReLU -> softmax(n) -> attention pool -> +centre_feat -> transposed out
//    One block per centroid, thread = channel; 32 neighbor values in registers.
// ---------------------------------------------------------------------------
__global__ __launch_bounds__(128) void attn_pool(const _Float16* __restrict__ Y,
                                                 const float* __restrict__ scale,
                                                 const float* __restrict__ shift,
                                                 const _Float16* __restrict__ np_final,
                                                 const _Float16* __restrict__ centre_feat,
                                                 float* __restrict__ out) {
  int bs = blockIdx.x;
  int c = threadIdx.x;
  int b = bs >> 10, s = bs & 1023;
  float sc = scale[c], sh = shift[c];
  float vals[NS];
  float mx = -3.4e38f;
#pragma unroll
  for (int n = 0; n < NS; ++n) {
    float v = fmaxf((float)Y[((size_t)(bs * NS + n)) * 128 + c] * sc + sh, 0.0f);
    vals[n] = v;
    mx = fmaxf(mx, v);
  }
  float se = 0.0f;
#pragma unroll
  for (int n = 0; n < NS; ++n) { float e = expf(vals[n] - mx); vals[n] = e; se += e; }
  float inv = 1.0f / se;
  float pooled = 0.0f;
#pragma unroll
  for (int n = 0; n < NS; ++n)
    pooled += vals[n] * inv * (float)np_final[((size_t)(bs * NS + n)) * 128 + c];
  float cf = (float)centre_feat[(size_t)bs * 128 + c];
  out[(size_t)(BB * 3 * SS) + ((size_t)b * 128 + c) * SS + s] = cf + pooled;
}

// ---------------------------------------------------------------------------
// 10. Pack f32 weight (Cout x Cin) -> f16 (Cout x Kpad), zero-padded K
// ---------------------------------------------------------------------------
__global__ void pack_w(const float* __restrict__ W, _Float16* __restrict__ Wp,
                       int Cin, int Kpad, int Cout) {
  int i = blockIdx.x * blockDim.x + threadIdx.x;
  if (i >= Cout * Kpad) return;
  int o = i / Kpad, k = i % Kpad;
  Wp[i] = (k < Cin) ? (_Float16)W[o * Cin + k] : (_Float16)0.0f;
}

// ---------------------------------------------------------------------------
// Host orchestration
// ---------------------------------------------------------------------------
extern "C" void kernel_launch(void* const* d_in, const int* in_sizes, int n_in,
                              void* d_out, int out_size, void* d_ws, size_t ws_size,
                              hipStream_t stream) {
  const float* xyz    = (const float*)d_in[0];
  const float* points = (const float*)d_in[1];
  const float* W_aug  = (const float*)d_in[2];
  const float* b_aug  = (const float*)d_in[3];
  const float* g_aug  = (const float*)d_in[4];
  const float* be_aug = (const float*)d_in[5];
  const float* W1     = (const float*)d_in[6];
  const float* b1     = (const float*)d_in[7];
  const float* g1     = (const float*)d_in[8];
  const float* be1    = (const float*)d_in[9];
  const float* W2     = (const float*)d_in[10];
  const float* b2     = (const float*)d_in[11];
  const float* g2     = (const float*)d_in[12];
  const float* be2    = (const float*)d_in[13];
  const float* W_laa  = (const float*)d_in[14];
  const float* b_laa  = (const float*)d_in[15];
  const float* g_laa  = (const float*)d_in[16];
  const float* be_laa = (const float*)d_in[17];
  float* out = (float*)d_out;

  // ---- workspace carve-up (aliased) ----
  char* base = (char*)d_ws;
  size_t off = 0;
  auto alloc = [&](size_t bytes) { size_t o = off; off += (bytes + 255) & ~(size_t)255; return o; };

  int*       fps_idx  = (int*)(base + alloc((size_t)P2 * 4));
  float*     new_xyz  = (float*)(base + alloc((size_t)P2 * 3 * 4));
  int*       knn_idx  = (int*)(base + alloc((size_t)P1 * 4));
  float*     grouped  = (float*)(base + alloc((size_t)P1 * 3 * 4));
  float*     gdist    = (float*)(base + alloc((size_t)P1 * 4));
  _Float16*  pts_t    = (_Float16*)(base + alloc((size_t)BB * NN * DD * 2));
  float*     stats    = (float*)(base + alloc(512 * 4)); // sum|sumsq|scale|shift (128 each)
  _Float16*  wp_aug   = (_Float16*)(base + alloc(32 * 32 * 2));
  _Float16*  wp1      = (_Float16*)(base + alloc(64 * 96 * 2));
  _Float16*  wp2      = (_Float16*)(base + alloc(128 * 64 * 2));
  _Float16*  wp_laa   = (_Float16*)(base + alloc(128 * 160 * 2));
  _Float16*  aug_in   = (_Float16*)(base + alloc((size_t)P1 * 32 * 2));
  _Float16*  fp_in    = (_Float16*)(base + alloc((size_t)P2 * 96 * 2));
  _Float16*  fp_h1    = (_Float16*)(base + alloc((size_t)P2 * 64 * 2));
  _Float16*  centre_f = (_Float16*)(base + alloc((size_t)P2 * 128 * 2));
  // np_in (P1*96*2) + h1 (P1*64*2) region == laa_in (P1*160*2), exact overlay
  size_t np_region    = alloc((size_t)P1 * 160 * 2);
  _Float16*  np_in    = (_Float16*)(base + np_region);
  _Float16*  h1       = (_Float16*)(base + np_region + (size_t)P1 * 96 * 2);
  _Float16*  laa_in   = (_Float16*)(base + np_region);
  _Float16*  np_final = (_Float16*)(base + alloc((size_t)P1 * 128 * 2));
  _Float16*  ybuf     = (_Float16*)(base + alloc((size_t)P1 * 128 * 2)); // pre-BN f16
  (void)ws_size; (void)in_sizes; (void)n_in; (void)out_size;

  float* sum   = stats;
  float* sumsq = stats + 128;
  float* scale = stats + 256;
  float* shift = stats + 384;

  auto zero_stats = [&]() {
    hipLaunchKernelGGL(zero_kernel, dim3(1), dim3(256), 0, stream, stats, 256);
  };
  auto finalize = [&](int P, int C, const float* gamma, const float* beta) {
    hipLaunchKernelGGL(bn_finalize, dim3(1), dim3(128), 0, stream,
                       sum, sumsq, gamma, beta, scale, shift, C, 1.0f / (float)P);
  };

  // ---- stage 0: weight packing ----
  hipLaunchKernelGGL(pack_w, dim3((32 * 32 + 255) / 256), dim3(256), 0, stream, W_aug, wp_aug, 10, 32, 32);
  hipLaunchKernelGGL(pack_w, dim3((64 * 96 + 255) / 256), dim3(256), 0, stream, W1, wp1, 96, 96, 64);
  hipLaunchKernelGGL(pack_w, dim3((128 * 64 + 255) / 256), dim3(256), 0, stream, W2, wp2, 64, 64, 128);
  hipLaunchKernelGGL(pack_w, dim3((128 * 160 + 255) / 256), dim3(256), 0, stream, W_laa, wp_laa, 135, 160, 128);

  // ---- stage 1: sampling / neighborhoods ----
  hipLaunchKernelGGL(fps_kernel, dim3(BB), dim3(1024), 0, stream, xyz, fps_idx);
  hipLaunchKernelGGL(gather_newxyz, dim3((P2 + 255) / 256), dim3(256), 0, stream,
                     xyz, fps_idx, new_xyz, out);
  hipLaunchKernelGGL(knn_kernel, dim3(P2), dim3(256), 0, stream, xyz, new_xyz, knn_idx);
  hipLaunchKernelGGL(transpose_points, dim3((BB * DD * NN + 255) / 256), dim3(256), 0, stream,
                     points, pts_t);
  hipLaunchKernelGGL(build_group, dim3((P1 + 255) / 256), dim3(256), 0, stream,
                     xyz, pts_t, new_xyz, knn_idx, grouped, gdist, aug_in, np_in);
  hipLaunchKernelGGL(fill_fp_points, dim3((P2 * DD + 255) / 256), dim3(256), 0, stream,
                     pts_t, fps_idx, fp_in);

  // ---- stage 2: aug conv_bn_relu (10->32); normalized result into np_in[...,64..95]
  zero_stats();
  hipLaunchKernelGGL(HIP_KERNEL_NAME(gemm_wmma<32, 2>), dim3(P1 / 128), dim3(256), 0, stream,
                     aug_in, wp_aug, b_aug, ybuf, sum, sumsq);
  finalize(P1, 32, g_aug, be_aug);
  hipLaunchKernelGGL(bn_apply_f16, dim3((int)(((size_t)P1 * 32 + 255) / 256)), dim3(256), 0, stream,
                     ybuf, scale, shift, np_in, P1, 32, 96, 64);
  hipLaunchKernelGGL(maxout_fp, dim3((P2 * 32 + 255) / 256), dim3(256), 0, stream, np_in, fp_in);

  // ---- stage 3: layer1 (96->64), fps branch then neighbor branch (separate BN stats) ----
  zero_stats();
  hipLaunchKernelGGL(HIP_KERNEL_NAME(gemm_wmma<96, 4>), dim3(P2 / 128), dim3(256), 0, stream,
                     fp_in, wp1, b1, ybuf, sum, sumsq);
  finalize(P2, 64, g1, be1);
  hipLaunchKernelGGL(bn_apply_f16, dim3((int)(((size_t)P2 * 64 + 255) / 256)), dim3(256), 0, stream,
                     ybuf, scale, shift, fp_h1, P2, 64, 64, 0);

  zero_stats();
  hipLaunchKernelGGL(HIP_KERNEL_NAME(gemm_wmma<96, 4>), dim3(P1 / 128), dim3(256), 0, stream,
                     np_in, wp1, b1, ybuf, sum, sumsq);
  finalize(P1, 64, g1, be1);
  hipLaunchKernelGGL(bn_apply_f16, dim3((int)(((size_t)P1 * 64 + 255) / 256)), dim3(256), 0, stream,
                     ybuf, scale, shift, h1, P1, 64, 64, 0);

  // ---- stage 4: layer2 (64->128) ----
  zero_stats();
  hipLaunchKernelGGL(HIP_KERNEL_NAME(gemm_wmma<64, 8>), dim3(P2 / 128), dim3(256), 0, stream,
                     fp_h1, wp2, b2, ybuf, sum, sumsq);
  finalize(P2, 128, g2, be2);
  hipLaunchKernelGGL(bn_apply_f16, dim3((int)(((size_t)P2 * 128 + 255) / 256)), dim3(256), 0, stream,
                     ybuf, scale, shift, centre_f, P2, 128, 128, 0);

  zero_stats();
  hipLaunchKernelGGL(HIP_KERNEL_NAME(gemm_wmma<64, 8>), dim3(P1 / 128), dim3(256), 0, stream,
                     h1, wp2, b2, ybuf, sum, sumsq);
  finalize(P1, 128, g2, be2);
  hipLaunchKernelGGL(bn_apply_f16, dim3((int)(((size_t)P1 * 128 + 255) / 256)), dim3(256), 0, stream,
                     ybuf, scale, shift, np_final, P1, 128, 128, 0);

  // ---- stage 5: LAA attention (135->128), softmax over neighbors, pool, add centre ----
  hipLaunchKernelGGL(build_laa, dim3(P1), dim3(160), 0, stream,
                     new_xyz, grouped, gdist, np_final, centre_f, laa_in);
  zero_stats();
  hipLaunchKernelGGL(HIP_KERNEL_NAME(gemm_wmma<160, 8>), dim3(P1 / 128), dim3(256), 0, stream,
                     laa_in, wp_laa, b_laa, ybuf, sum, sumsq);
  finalize(P1, 128, g_laa, be_laa);
  hipLaunchKernelGGL(attn_pool, dim3(P2), dim3(128), 0, stream,
                     ybuf, scale, shift, np_final, centre_f, out);
}
